// Attention_8443905704227
// MI455X (gfx1250) — matrix operand order repaired and checked by reading, not compiled, not run on hardware
//
#include <hip/hip_runtime.h>

// ---------------------------------------------------------------------------
// CDNA5 (gfx1250) fused attention: fp32 in/out, bf16 WMMA compute.
//   B=4, N=2048, C=768, H=12, D=64
//   k1: qkv = x @ qkv_w^T           -> Q,K,V bf16 workspace [B,H,N,D]
//   k2: flash attention, double-buffered async K/V tiles (ASYNCcnt),
//       online softmax with LDS score relayout (C->A) and 1 shfl per reduce
//   k3: out = ctx @ proj_w^T        -> fp32 d_out
// All matmuls use v_wmma_f32_16x16x32_bf16 (wave32, f32 accumulate).
// ---------------------------------------------------------------------------

typedef __attribute__((ext_vector_type(16))) __bf16 v16bf;
typedef __attribute__((ext_vector_type(8)))  __bf16 v8bf;
typedef __attribute__((ext_vector_type(8)))  float  v8f;
typedef __attribute__((ext_vector_type(4)))  float  v4f;

#define WMMA_BF16(a, b, c) \
  __builtin_amdgcn_wmma_f32_16x16x32_bf16(false, (a), false, (b), (short)0, (c), false, false)

static constexpr int Bb = 4;
static constexpr int Nn = 2048;
static constexpr int Cc = 768;
static constexpr int Hh = 12;
static constexpr int Dd = 64;

// ---------------------------------------------------------------------------
// GEMM: Out[M,F] = A[M,K] * W[F,K]^T   (A,W fp32 -> bf16 on stage)
// 256 threads = 8 waves, 64x64 tile, K-step 32, b128 staging.
// MODE 0: scatter into Q/K/V bf16 [B,H,N,D].   MODE 1: fp32 Out[M,F].
// ---------------------------------------------------------------------------
static constexpr int LDK = 40;  // bf16 k-stride in LDS (80 B rows, 16B aligned)

template <int MODE>
__global__ __launch_bounds__(256) void gemm_bf16_wmma(
    const float* __restrict__ A, const float* __restrict__ W,
    __bf16* __restrict__ qout, __bf16* __restrict__ kout, __bf16* __restrict__ vout,
    float* __restrict__ out, int M, int K, int F) {
  __shared__ __bf16 As[64 * LDK];
  __shared__ __bf16 Ws[64 * LDK];

  const int tid = threadIdx.x;
  const int lane = tid & 31, wave = tid >> 5;
  const int laneHi = lane >> 4, lane16 = lane & 15;
  const int m0 = blockIdx.y * 64, f0 = blockIdx.x * 64;
  const int sm = wave >> 1, sn0 = (wave & 1) * 2;  // 4x4 subtiles, 2 per wave
  // staging: thread owns 8 consecutive k in one row (one b128 pair + b128 LDS store)
  const int srow = tid >> 2, skc = (tid & 3) * 8;
  const float* Ag = A + (size_t)(m0 + srow) * K + skc;
  const float* Wg = W + (size_t)(f0 + srow) * K + skc;

  v8f acc0 = {}, acc1 = {};

  for (int k0 = 0; k0 < K; k0 += 32) {
    __syncthreads();
    v4f a0 = *(const v4f*)(Ag + k0);
    v4f a1 = *(const v4f*)(Ag + k0 + 4);
    v4f w0 = *(const v4f*)(Wg + k0);
    v4f w1 = *(const v4f*)(Wg + k0 + 4);
    if (k0 + 32 < K) {  // pull next K-tile toward L2/WGP$
      __builtin_prefetch(Ag + k0 + 32, 0, 0);
      __builtin_prefetch(Wg + k0 + 32, 0, 0);
    }
    v8bf ap, wp;
#pragma unroll
    for (int j = 0; j < 4; ++j) {
      ap[j] = (__bf16)a0[j]; ap[4 + j] = (__bf16)a1[j];
      wp[j] = (__bf16)w0[j]; wp[4 + j] = (__bf16)w1[j];
    }
    *(v8bf*)&As[srow * LDK + skc] = ap;
    *(v8bf*)&Ws[srow * LDK + skc] = wp;
    __syncthreads();

    // A fragment (16x32 bf16, ISA §7.12.2 layout)
    v16bf a;
    const int arow = sm * 16 + lane16;
#pragma unroll
    for (int e = 0; e < 16; ++e) {
      int ka = (e & 7) + ((e >> 3) << 4) + (laneHi << 3);
      a[e] = As[arow * LDK + ka];
    }
    // B fragments (32x16): B[k, n=f] = W[f, k]
    v16bf b0, b1;
    const int frow0 = sn0 * 16 + lane16;
#pragma unroll
    for (int e = 0; e < 16; ++e) {
      int kb = (e & 7) + ((e >> 3) << 4) + (laneHi << 3);
      b0[e] = Ws[frow0 * LDK + kb];
      b1[e] = Ws[(frow0 + 16) * LDK + kb];
    }
    acc0 = WMMA_BF16(a, b0, acc0);
    acc1 = WMMA_BF16(a, b1, acc1);
  }

  // Epilogue. C-layout: row = r + 8*laneHi, col = lane16.
#pragma unroll
  for (int s = 0; s < 2; ++s) {
    v8f acc = s ? acc1 : acc0;
    const int f = f0 + (sn0 + s) * 16 + lane16;
    if constexpr (MODE == 0) {
      const int which = f / Cc;  // 0=q 1=k 2=v
      const int hd = f % Cc;
      const int h = hd >> 6, d = hd & 63;
      __bf16* dst = (which == 0) ? qout : (which == 1) ? kout : vout;
#pragma unroll
      for (int r = 0; r < 8; ++r) {
        int m = m0 + sm * 16 + r + 8 * laneHi;
        int b = m >> 11, n = m & (Nn - 1);
        dst[(((size_t)(b * Hh + h)) * Nn + n) * Dd + d] = (__bf16)acc[r];
      }
    } else {
#pragma unroll
      for (int r = 0; r < 8; ++r) {
        int m = m0 + sm * 16 + r + 8 * laneHi;
        out[(size_t)m * F + f] = acc[r];
      }
    }
  }
}

// ---------------------------------------------------------------------------
// Flash attention. Block = 8 waves on one (b,h), 128 queries (16 per wave).
// K/V tiles of 32 keys staged by global_load_async_to_lds_b128, double-buffered.
// ---------------------------------------------------------------------------
static constexpr int LDD = 72;  // bf16 d-stride for K/V LDS rows (144 B, 16B aligned)

__global__ __launch_bounds__(256) void flash_attn_wmma(
    const __bf16* __restrict__ Qg, const __bf16* __restrict__ Kg,
    const __bf16* __restrict__ Vg, float* __restrict__ ctx) {
  __shared__ __bf16 Klds[2][32 * LDD];
  __shared__ __bf16 Vlds[2][32 * LDD];
  __shared__ float  Sbuf[8][16 * 32];  // per-wave score relayout scratch
  __shared__ float  Xbuf[8][16];       // per-wave alpha / inv-l row broadcast

  const int tid = threadIdx.x;
  const int lane = tid & 31, wave = tid >> 5;
  const int laneHi = lane >> 4, lane16 = lane & 15;
  const int bh = blockIdx.y;  // b*H + h
  const int qbase = blockIdx.x * 128 + wave * 16;
  const size_t base = (size_t)bh * Nn * Dd;

  // Q fragments: 16 queries x 64 d, two K=32 chunks (A layout), loaded once.
  v16bf qf[2];
  {
    const size_t qoff = base + (size_t)(qbase + lane16) * Dd;
#pragma unroll
    for (int c2 = 0; c2 < 2; ++c2)
#pragma unroll
      for (int e = 0; e < 16; ++e) {
        int dd = 32 * c2 + (e & 7) + ((e >> 3) << 4) + (laneHi << 3);
        qf[c2][e] = Qg[qoff + dd];
      }
  }

  v8f zero8 = {};
  v8f accO[4] = {zero8, zero8, zero8, zero8};  // 16 x 64 fp32 context accum
  float mrow = -3.0e38f, lrow = 0.0f;          // per-lane (row = lane16)

  // async staging: one 16-byte chunk per lane per tensor
  const int skey = tid >> 3, sdp = (tid & 7) * 8;
  const __bf16* Kbh = Kg + base + (size_t)skey * Dd + sdp;
  const __bf16* Vbh = Vg + base + (size_t)skey * Dd + sdp;
  const uint32_t kldsoff = (uint32_t)(uintptr_t)&Klds[0][skey * LDD + sdp];
  const uint32_t vldsoff = (uint32_t)(uintptr_t)&Vlds[0][skey * LDD + sdp];
  const uint32_t bufstep = (uint32_t)(32 * LDD * sizeof(__bf16));

  auto issue = [&](int kt, int buf) {
    uint64_t kg = (uint64_t)(uintptr_t)(Kbh + (size_t)kt * Dd);
    uint64_t vg = (uint64_t)(uintptr_t)(Vbh + (size_t)kt * Dd);
    uint32_t ko = kldsoff + buf * bufstep;
    uint32_t vo = vldsoff + buf * bufstep;
    asm volatile(
        "global_load_async_to_lds_b128 %0, %2, off\n\t"
        "global_load_async_to_lds_b128 %1, %3, off"
        :: "v"(ko), "v"(vo), "v"(kg), "v"(vg) : "memory");
  };

  constexpr int NT = Nn / 32;
  issue(0, 0);
  for (int it = 0; it < NT; ++it) {
    const int buf = it & 1;
    if (it + 1 < NT) {
      issue((it + 1) * 32, buf ^ 1);
      asm volatile("s_wait_asynccnt 2" ::: "memory");  // tile `it` landed
    } else {
      asm volatile("s_wait_asynccnt 0" ::: "memory");
    }
    __syncthreads();

    const __bf16* Kt = Klds[buf];
    const __bf16* Vt = Vlds[buf];

    // S = Q K^T : two 16x16 score tiles, each accumulated over d in 2 WMMAs
    v8f s0 = {}, s1 = {};
#pragma unroll
    for (int c2 = 0; c2 < 2; ++c2) {
      v16bf bk0, bk1;
#pragma unroll
      for (int e = 0; e < 16; ++e) {
        int dd = 32 * c2 + (e & 7) + ((e >> 3) << 4) + (laneHi << 3);
        bk0[e] = Kt[lane16 * LDD + dd];
        bk1[e] = Kt[(16 + lane16) * LDD + dd];
      }
      s0 = WMMA_BF16(qf[c2], bk0, s0);
      s1 = WMMA_BF16(qf[c2], bk1, s1);
    }

    // C -> A relayout of scaled scores through wave-private LDS
    float* Sw = Sbuf[wave];
#pragma unroll
    for (int r = 0; r < 8; ++r) {
      int prow = r + 8 * laneHi;
      Sw[prow * 32 + lane16]      = s0[r] * 0.125f;  // 1/sqrt(64)
      Sw[prow * 32 + 16 + lane16] = s1[r] * 0.125f;
    }
    asm volatile("s_wait_dscnt 0" ::: "memory");

    // lane now owns row lane16 with 16 of the 32 key-columns (A layout)
    float sv[16];
#pragma unroll
    for (int e = 0; e < 16; ++e) {
      int kk = (e & 7) + ((e >> 3) << 4) + (laneHi << 3);
      sv[e] = Sw[lane16 * 32 + kk];
    }
    float mt = sv[0];
#pragma unroll
    for (int e = 1; e < 16; ++e) mt = fmaxf(mt, sv[e]);
    mt = fmaxf(mt, __shfl_xor(mt, 16, 32));  // combine key halves: 1 bpermute
    float mnew = fmaxf(mrow, mt);
    float alpha = __expf(mrow - mnew);
    mrow = mnew;

    v16bf pf;
    float rs = 0.0f;
#pragma unroll
    for (int e = 0; e < 16; ++e) {
      float p = __expf(sv[e] - mnew);
      rs += p;
      pf[e] = (__bf16)p;  // A-fragment of P, directly
    }
    rs += __shfl_xor(rs, 16, 32);
    lrow = lrow * alpha + rs;

    // broadcast per-row alpha to C-layout consumers
    Xbuf[wave][lane16] = alpha;  // both halves store identical values
    asm volatile("s_wait_dscnt 0" ::: "memory");
    float al[8];
#pragma unroll
    for (int r = 0; r < 8; ++r) al[r] = Xbuf[wave][r + 8 * laneHi];

    // O = O*alpha + P V  (four 16-col slabs of D=64)
#pragma unroll
    for (int t = 0; t < 4; ++t) {
      v16bf bv;
#pragma unroll
      for (int e = 0; e < 16; ++e) {
        int kk = (e & 7) + ((e >> 3) << 4) + (laneHi << 3);
        bv[e] = Vt[kk * LDD + t * 16 + lane16];
      }
      v8f cs;
#pragma unroll
      for (int r = 0; r < 8; ++r) cs[r] = accO[t][r] * al[r];
      accO[t] = WMMA_BF16(pf, bv, cs);
    }
    __syncthreads();  // protect buffer `buf` before it is re-issued
  }

  // finalize: 1/l per row broadcast, write ctx as [B,N,C] fp32
  Xbuf[wave][lane16] = 1.0f / lrow;
  asm volatile("s_wait_dscnt 0" ::: "memory");
  float inv[8];
#pragma unroll
  for (int r = 0; r < 8; ++r) inv[r] = Xbuf[wave][r + 8 * laneHi];

  const int b = bh / Hh, h = bh % Hh;
#pragma unroll
  for (int r = 0; r < 8; ++r) {
    int qr = qbase + r + 8 * laneHi;
    size_t rowoff = ((size_t)(b * Nn + qr)) * Cc + h * Dd;
#pragma unroll
    for (int t = 0; t < 4; ++t) ctx[rowoff + t * 16 + lane16] = accO[t][r] * inv[r];
  }
}

// ---------------------------------------------------------------------------
extern "C" void kernel_launch(void* const* d_in, const int* in_sizes, int n_in,
                              void* d_out, int out_size, void* d_ws, size_t ws_size,
                              hipStream_t stream) {
  const float* x      = (const float*)d_in[0];  // [B,N,C]
  const float* qkv_w  = (const float*)d_in[1];  // [3C,C]
  const float* proj_w = (const float*)d_in[2];  // [C,C]
  float* out = (float*)d_out;                   // [B,N,C] fp32

  // workspace: Q,K,V bf16 [B,H,N,D] + ctx fp32 [B,N,C]  (~63 MB)
  const size_t qkv_elems = (size_t)Bb * Hh * Nn * Dd;
  __bf16* Qws = (__bf16*)d_ws;
  __bf16* Kws = Qws + qkv_elems;
  __bf16* Vws = Kws + qkv_elems;
  float* ctx = (float*)(Vws + qkv_elems);

  dim3 blk(256);
  gemm_bf16_wmma<0><<<dim3(3 * Cc / 64, Bb * Nn / 64), blk, 0, stream>>>(
      x, qkv_w, Qws, Kws, Vws, nullptr, Bb * Nn, Cc, 3 * Cc);
  flash_attn_wmma<<<dim3(Nn / 128, Bb * Hh), blk, 0, stream>>>(Qws, Kws, Vws, ctx);
  gemm_bf16_wmma<1><<<dim3(Cc / 64, Bb * Nn / 64), blk, 0, stream>>>(
      ctx, proj_w, nullptr, nullptr, nullptr, out, Bb * Nn, Cc, Cc);
}
